// Simba_75634374082997
// MI455X (gfx1250) — matrix operand order repaired and checked by reading, not compiled, not run on hardware
//
#include <hip/hip_runtime.h>

// ---------------------------------------------------------------------------
// Simba (Mamba) block forward for MI455X / gfx1250 (wave32, WMMA).
// Strategy: bf16 WMMA (v_wmma_f32_16x16x32_bf16, f32 accum) for all four
// projections (memory-bound at 23.3 TB/s), fp32 everywhere precision matters
// (LayerNorm, conv, softplus, recurrent scan). Heavy epilogue fusion to
// minimize HBM round-trips. Scan parallelizes the state dim across lanes
// (wave32 shuffle reduction) for full-machine occupancy.
// ---------------------------------------------------------------------------

typedef __attribute__((ext_vector_type(16))) __bf16 v16bf;
typedef __attribute__((ext_vector_type(8)))  __bf16 v8bf;
typedef __attribute__((ext_vector_type(8)))  float  v8f;

constexpr int B_SZ    = 2;
constexpr int L       = 2048;
constexpr int DIM     = 1024;
constexpr int D_INNER = 2048;
constexpr int D_STATE = 64;
constexpr int D_CONV  = 64;
constexpr int DT_RANK = 64;
constexpr int NT      = B_SZ * L;          // 4096 tokens
constexpr int XDBL_W  = DT_RANK + 2 * D_STATE;  // 192
constexpr float LN_EPS = 1e-5f;

// ---------------------------------------------------------------------------
// WMMA fragment loaders (ISA 7.12.2 layouts, wave32).
// A 16x32 bf16 (MxK): lane = half*16 + m; holds K = half*8+{0..7} and
// 16+half*8+{0..7}  -> two contiguous 16B chunks -> global_load_b128.
// B 32x16 bf16 (KxN) built from W stored (N,K) row-major: lane = kh*16 + n;
// holds W[n][k0 + kh*8 + 0..7] and W[n][k0 + 16 + kh*8 + 0..7] (same K as A).
// ---------------------------------------------------------------------------
__device__ __forceinline__ v16bf load_a_frag(const __bf16* __restrict__ A,
                                             int lda, int m0, int k0) {
  const int lane = threadIdx.x & 31;
  const int m    = m0 + (lane & 15);
  const int half = lane >> 4;
  const __bf16* p = A + (size_t)m * lda + k0 + half * 8;
  v8bf lo = *(const v8bf*)p;          // K = half*8 .. +7
  v8bf hi = *(const v8bf*)(p + 16);   // K = 16+half*8 .. +7
  v16bf a;
#pragma unroll
  for (int i = 0; i < 8; ++i) { a[i] = lo[i]; a[8 + i] = hi[i]; }
  return a;
}

__device__ __forceinline__ v16bf load_b_frag(const __bf16* __restrict__ W,
                                             int ldw, int n0, int k0) {
  const int lane = threadIdx.x & 31;
  const int n    = n0 + (lane & 15);
  const int kh   = lane >> 4;
  const __bf16* p = W + (size_t)n * ldw + k0 + kh * 8;
  v8bf lo = *(const v8bf*)p;
  v8bf hi = *(const v8bf*)(p + 16);
  v16bf b;
#pragma unroll
  for (int i = 0; i < 8; ++i) { b[i] = lo[i]; b[8 + i] = hi[i]; }
  return b;
}

// ---------------------------------------------------------------------------
// Generic bf16 WMMA GEMM: C[M,N] = A[M,K] (row-major) * W[N,K]^T (row-major).
// Block = 256 threads = 8 waves arranged 4(M) x 2(N); each wave owns a
// 32x32 output patch (2x2 of 16x16 WMMA tiles). Block tile = 128M x 64N.
// mode 0: plain store          (out0, ldc=N)
// mode 1: split halves         (cols<D_INNER -> out0=u, else out1=res)
// mode 2: +bias, softplus      (aux = bias[N], out0=delta)
// mode 3: +residual            (aux = x[M,N], out0=out2)
// ---------------------------------------------------------------------------
__global__ __launch_bounds__(256)
void gemm_bf16_wmma_kernel(const __bf16* __restrict__ A, int lda,
                           const __bf16* __restrict__ W, int ldw,
                           int M, int N, int K, int mode,
                           const float* __restrict__ aux,
                           float* __restrict__ out0,
                           float* __restrict__ out1) {
  const int wave = threadIdx.x >> 5;
  const int lane = threadIdx.x & 31;
  const int wm = wave & 3;
  const int wn = wave >> 2;
  const int m_base = blockIdx.x * 128 + wm * 32;
  const int n_base = blockIdx.y * 64 + wn * 32;

  v8f acc00 = {}, acc01 = {}, acc10 = {}, acc11 = {};

  for (int k = 0; k < K; k += 32) {
    // prefetch next k-tile into GL2 (global_prefetch_b8)
    if (k + 32 < K) {
      __builtin_prefetch(A + (size_t)(m_base + (lane & 15)) * lda + k + 32, 0, 3);
      __builtin_prefetch(W + (size_t)(n_base + (lane & 15)) * ldw + k + 32, 0, 3);
    }
    v16bf a0 = load_a_frag(A, lda, m_base,      k);
    v16bf a1 = load_a_frag(A, lda, m_base + 16, k);
    v16bf b0 = load_b_frag(W, ldw, n_base,      k);
    v16bf b1 = load_b_frag(W, ldw, n_base + 16, k);
    acc00 = __builtin_amdgcn_wmma_f32_16x16x32_bf16(false, a0, false, b0,
                                                    (short)0, acc00, false, false);
    acc01 = __builtin_amdgcn_wmma_f32_16x16x32_bf16(false, a0, false, b1,
                                                    (short)0, acc01, false, false);
    acc10 = __builtin_amdgcn_wmma_f32_16x16x32_bf16(false, a1, false, b0,
                                                    (short)0, acc10, false, false);
    acc11 = __builtin_amdgcn_wmma_f32_16x16x32_bf16(false, a1, false, b1,
                                                    (short)0, acc11, false, false);
  }

  // C 16x16 f32 layout: VGPR r, lanes0-15 -> M=r, lanes16-31 -> M=8+r; N=lane&15
  const int col   = lane & 15;
  const int rbase = (lane >> 4) * 8;
  v8f accs[2][2] = { { acc00, acc01 }, { acc10, acc11 } };
#pragma unroll
  for (int i = 0; i < 2; ++i) {
#pragma unroll
    for (int j = 0; j < 2; ++j) {
      const int m0 = m_base + 16 * i;
      const int n0 = n_base + 16 * j;
#pragma unroll
      for (int r = 0; r < 8; ++r) {
        const int m = m0 + rbase + r;
        const int n = n0 + col;
        float v = accs[i][j][r];
        if (mode == 0) {
          out0[(size_t)m * N + n] = v;
        } else if (mode == 1) {
          if (n < D_INNER) out0[(size_t)m * D_INNER + n] = v;
          else             out1[(size_t)m * D_INNER + (n - D_INNER)] = v;
        } else if (mode == 2) {
          float t = v + aux[n];
          // numerically stable softplus
          float sp = (t > 20.f) ? t
                   : (t < -20.f) ? __expf(t)
                   : log1pf(__expf(t));
          out0[(size_t)m * N + n] = sp;
        } else {
          out0[(size_t)m * N + n] = v + aux[(size_t)m * N + n];
        }
      }
    }
  }
}

// ---------------------------------------------------------------------------
// LayerNorm (weight=1,bias=0) over DIM, emitting bf16 for the in_proj GEMM.
// One block (256 threads / 8 waves) per token row.
// ---------------------------------------------------------------------------
__global__ __launch_bounds__(256)
void ln_to_bf16_kernel(const float* __restrict__ x, __bf16* __restrict__ out) {
  const int row = blockIdx.x;
  const float* xr = x + (size_t)row * DIM;
  float s = 0.f, s2 = 0.f;
  for (int i = threadIdx.x; i < DIM; i += 256) {
    float v = xr[i];
    s += v; s2 += v * v;
  }
#pragma unroll
  for (int o = 16; o > 0; o >>= 1) {
    s  += __shfl_xor(s, o, 32);
    s2 += __shfl_xor(s2, o, 32);
  }
  __shared__ float ss[8], ss2[8];
  if ((threadIdx.x & 31) == 0) { ss[threadIdx.x >> 5] = s; ss2[threadIdx.x >> 5] = s2; }
  __syncthreads();
  s = 0.f; s2 = 0.f;
#pragma unroll
  for (int w = 0; w < 8; ++w) { s += ss[w]; s2 += ss2[w]; }
  const float mean = s * (1.f / DIM);
  const float var  = s2 * (1.f / DIM) - mean * mean;
  const float rstd = rsqrtf(var + LN_EPS);
  for (int i = threadIdx.x; i < DIM; i += 256)
    out[(size_t)row * DIM + i] = (__bf16)((xr[i] - mean) * rstd);
}

// Final: d_out = LN(out2) + out2 (fp32)
__global__ __launch_bounds__(256)
void ln_add_kernel(const float* __restrict__ x, float* __restrict__ out) {
  const int row = blockIdx.x;
  const float* xr = x + (size_t)row * DIM;
  float s = 0.f, s2 = 0.f;
  for (int i = threadIdx.x; i < DIM; i += 256) {
    float v = xr[i];
    s += v; s2 += v * v;
  }
#pragma unroll
  for (int o = 16; o > 0; o >>= 1) {
    s  += __shfl_xor(s, o, 32);
    s2 += __shfl_xor(s2, o, 32);
  }
  __shared__ float ss[8], ss2[8];
  if ((threadIdx.x & 31) == 0) { ss[threadIdx.x >> 5] = s; ss2[threadIdx.x >> 5] = s2; }
  __syncthreads();
  s = 0.f; s2 = 0.f;
#pragma unroll
  for (int w = 0; w < 8; ++w) { s += ss[w]; s2 += ss2[w]; }
  const float mean = s * (1.f / DIM);
  const float var  = s2 * (1.f / DIM) - mean * mean;
  const float rstd = rsqrtf(var + LN_EPS);
  for (int i = threadIdx.x; i < DIM; i += 256) {
    float v = xr[i];
    out[(size_t)row * DIM + i] = (v - mean) * rstd + v;
  }
}

// ---------------------------------------------------------------------------
// Causal depthwise conv (D_CONV=64 taps) + bias + SiLU, dual fp32/bf16 store.
// Block = 256 threads = (64 channels x 4 l); LDS tile of 67 input rows.
// Weight tile padded to 65 cols to avoid 64-way LDS bank conflicts.
// ---------------------------------------------------------------------------
__global__ __launch_bounds__(256)
void conv_silu_kernel(const float* __restrict__ u,
                      const float* __restrict__ cw,
                      const float* __restrict__ cb,
                      float* __restrict__ uc,
                      __bf16* __restrict__ ucb) {
  __shared__ float su[67 * 64];
  __shared__ float sw[64 * 65];
  const int b   = blockIdx.z;
  const int ch0 = blockIdx.y * 64;
  const int l0  = blockIdx.x * 4;
  const int tx  = threadIdx.x & 63;   // channel within tile
  const int ty  = threadIdx.x >> 6;   // l within tile (0..3)

  for (int i = threadIdx.x; i < 64 * 64; i += 256) {
    int c = i >> 6, t = i & 63;
    sw[c * 65 + t] = cw[(size_t)(ch0 + c) * D_CONV + t];
  }
  for (int i = threadIdx.x; i < 67 * 64; i += 256) {
    int r = i >> 6, c = i & 63;
    int l = l0 - (D_CONV - 1) + r;
    su[r * 64 + c] = (l >= 0)
        ? u[((size_t)b * L + l) * D_INNER + ch0 + c] : 0.f;
  }
  __syncthreads();

  const int ch = ch0 + tx;
  float acc = cb[ch];
#pragma unroll
  for (int t = 0; t < D_CONV; ++t)
    acc = fmaf(su[(ty + t) * 64 + tx], sw[tx * 65 + t], acc);
  const float sv = acc / (1.f + __expf(-acc));   // SiLU
  const size_t idx = ((size_t)b * L + (l0 + ty)) * D_INNER + ch;
  uc[idx]  = sv;
  ucb[idx] = (__bf16)sv;
}

// ---------------------------------------------------------------------------
// Extract dt columns (first 64 of the 192-wide x_dbl) as bf16 A for dt GEMM.
// ---------------------------------------------------------------------------
__global__ __launch_bounds__(256)
void extract_dt_kernel(const float* __restrict__ xdbl, __bf16* __restrict__ dtb) {
  const int i = blockIdx.x * 256 + threadIdx.x;
  if (i >= NT * DT_RANK) return;
  const int m = i >> 6, r = i & 63;
  dtb[i] = (__bf16)xdbl[(size_t)m * XDBL_W + r];
}

// f32 -> bf16 conversion (weights / activations)
__global__ __launch_bounds__(256)
void cvt_bf16_kernel(const float* __restrict__ src, __bf16* __restrict__ dst, int n) {
  const int i = blockIdx.x * 256 + threadIdx.x;
  if (i < n) dst[i] = (__bf16)src[i];
}

// ---------------------------------------------------------------------------
// Selective scan v2: state dimension spread across lanes.
// Each channel uses 16 lanes; each lane owns 4 states (n = nl + 16q, stride-16
// so B_t/C_t reads coalesce). 2 channels per wave, 16 channels per 256-thread
// block -> grid (D_INNER/16, B) = 2048 waves (full-machine occupancy) vs 128
// in the thread-per-channel version. Per-step serial work per lane: 4 exp +
// 12 FMA + 4 xor-shuffles (y reduction stays inside each 16-lane group on
// wave32). Skip term + SiLU gate + bf16 emit fused; lane 0 of each group
// writes. B/C for the whole sequence is 2 MB -> L2-resident re-reads.
// ---------------------------------------------------------------------------
__global__ __launch_bounds__(256)
void scan_gate_kernel(const float* __restrict__ uc,
                      const float* __restrict__ delta,
                      const float* __restrict__ xdbl,
                      const float* __restrict__ A_log,
                      const float* __restrict__ res,
                      const float* __restrict__ Dvec,
                      __bf16* __restrict__ yb) {
  const int b    = blockIdx.y;
  const int wave = threadIdx.x >> 5;
  const int lane = threadIdx.x & 31;
  const int sub  = lane >> 4;          // which of the wave's 2 channels
  const int nl   = lane & 15;          // state lane within channel group
  const int ch   = blockIdx.x * 16 + wave * 2 + sub;

  float Arow[4], s[4];
#pragma unroll
  for (int q = 0; q < 4; ++q) {
    Arow[q] = -__expf(A_log[(size_t)ch * D_STATE + nl + 16 * q]);
    s[q] = 0.f;
  }
  const float Dch = Dvec[ch];

  const float* Bbase = xdbl + DT_RANK;
  const float* Cbase = xdbl + DT_RANK + D_STATE;

  for (int l = 0; l < L; ++l) {
    const size_t tok = (size_t)b * L + l;
    const size_t ci  = tok * D_INNER + ch;
    const float u_t = uc[ci];      // uniform across the 16 lanes (L0 bcast)
    const float dt  = delta[ci];
    const float dtu = dt * u_t;
    const float* Bp = Bbase + tok * XDBL_W;
    const float* Cp = Cbase + tok * XDBL_W;
    float yp = 0.f;
#pragma unroll
    for (int q = 0; q < 4; ++q) {
      const float dA = __expf(dt * Arow[q]);
      s[q] = fmaf(s[q], dA, dtu * Bp[nl + 16 * q]);
      yp = fmaf(s[q], Cp[nl + 16 * q], yp);
    }
    // y_t = sum over the channel's 16 lanes (xor masks stay within group)
#pragma unroll
    for (int m = 8; m >= 1; m >>= 1)
      yp += __shfl_xor(yp, m, 32);
    if (nl == 0) {
      float yv = fmaf(u_t, Dch, yp);
      const float r = res[ci];
      yv *= r / (1.f + __expf(-r));   // SiLU gate
      yb[ci] = (__bf16)yv;
    }
  }
}

// ---------------------------------------------------------------------------
// Host-side orchestration.
// ---------------------------------------------------------------------------
extern "C" void kernel_launch(void* const* d_in, const int* in_sizes, int n_in,
                              void* d_out, int out_size, void* d_ws, size_t ws_size,
                              hipStream_t stream) {
  const float* x         = (const float*)d_in[0];  // (B,L,DIM)
  const float* in_proj_w = (const float*)d_in[1];  // (2*D_INNER, DIM)
  const float* conv_w    = (const float*)d_in[2];  // (D_INNER,1,D_CONV)
  const float* conv_b    = (const float*)d_in[3];  // (D_INNER)
  const float* x_proj_w  = (const float*)d_in[4];  // (192, D_INNER)
  const float* dt_proj_w = (const float*)d_in[5];  // (D_INNER, DT_RANK)
  const float* dt_proj_b = (const float*)d_in[6];  // (D_INNER)
  const float* A_log     = (const float*)d_in[7];  // (D_INNER, D_STATE)
  const float* Dvec      = (const float*)d_in[8];  // (D_INNER)
  const float* out_proj_w= (const float*)d_in[9];  // (DIM, D_INNER)
  float* out = (float*)d_out;

  // Workspace carve-up (256B aligned)
  char* w = (char*)d_ws;
  size_t o = 0;
  auto carve = [&](size_t bytes) {
    void* p = w + o;
    o = (o + bytes + 255) & ~(size_t)255;
    return p;
  };
  __bf16* xnb  = (__bf16*)carve((size_t)NT * DIM * 2);            // LN(x) bf16
  __bf16* winb = (__bf16*)carve((size_t)2 * D_INNER * DIM * 2);   // in_proj_w bf16
  __bf16* xpwb = (__bf16*)carve((size_t)XDBL_W * D_INNER * 2);    // x_proj_w bf16
  __bf16* dtwb = (__bf16*)carve((size_t)D_INNER * DT_RANK * 2);   // dt_proj_w bf16
  __bf16* opwb = (__bf16*)carve((size_t)DIM * D_INNER * 2);       // out_proj_w bf16
  float*  u    = (float*)carve((size_t)NT * D_INNER * 4);         // pre-conv u
  float*  res  = (float*)carve((size_t)NT * D_INNER * 4);         // gate branch
  float*  uc   = (float*)carve((size_t)NT * D_INNER * 4);         // conv+SiLU f32
  __bf16* ucb  = (__bf16*)carve((size_t)NT * D_INNER * 2);        // conv+SiLU bf16
  float*  xdbl = (float*)carve((size_t)NT * XDBL_W * 4);          // x_proj out
  __bf16* dtb  = (__bf16*)carve((size_t)NT * DT_RANK * 2);        // dt bf16
  __bf16* yb   = (__bf16*)carve((size_t)NT * D_INNER * 2);        // gated y bf16
  float*  out2 = (float*)carve((size_t)NT * DIM * 4);             // residual+mamba
  float*  delta = u;  // u is dead after conv; reuse its 32MB for delta

  auto cvt = [&](const float* src, __bf16* dst, int n) {
    cvt_bf16_kernel<<<(n + 255) / 256, 256, 0, stream>>>(src, dst, n);
  };

  // 0) weights -> bf16
  cvt(in_proj_w,  winb, 2 * D_INNER * DIM);
  cvt(x_proj_w,   xpwb, XDBL_W * D_INNER);
  cvt(dt_proj_w,  dtwb, D_INNER * DT_RANK);
  cvt(out_proj_w, opwb, DIM * D_INNER);

  // 1) xn = LN(x) -> bf16
  ln_to_bf16_kernel<<<NT, 256, 0, stream>>>(x, xnb);

  // 2) in_proj: (4096x1024)x(1024x4096) -> split u | res
  gemm_bf16_wmma_kernel<<<dim3(NT / 128, (2 * D_INNER) / 64), 256, 0, stream>>>(
      xnb, DIM, winb, DIM, NT, 2 * D_INNER, DIM, /*mode=*/1, nullptr, u, res);

  // 3) causal depthwise conv + SiLU  (u -> uc, ucb)
  conv_silu_kernel<<<dim3(L / 4, D_INNER / 64, B_SZ), 256, 0, stream>>>(
      u, conv_w, conv_b, uc, ucb);

  // 4) x_proj: (4096x2048)x(2048x192) -> xdbl
  gemm_bf16_wmma_kernel<<<dim3(NT / 128, XDBL_W / 64), 256, 0, stream>>>(
      ucb, D_INNER, xpwb, D_INNER, NT, XDBL_W, D_INNER, /*mode=*/0, nullptr,
      xdbl, nullptr);

  // 5) dt slice -> bf16
  extract_dt_kernel<<<(NT * DT_RANK + 255) / 256, 256, 0, stream>>>(xdbl, dtb);

  // 6) dt_proj + bias + softplus -> delta (reuses u's storage)
  gemm_bf16_wmma_kernel<<<dim3(NT / 128, D_INNER / 64), 256, 0, stream>>>(
      dtb, DT_RANK, dtwb, DT_RANK, NT, D_INNER, DT_RANK, /*mode=*/2, dt_proj_b,
      delta, nullptr);

  // 7) selective scan (lane-parallel state), fused skip + gate -> yb (bf16)
  scan_gate_kernel<<<dim3(D_INNER / 16, B_SZ), 256, 0, stream>>>(
      uc, delta, xdbl, A_log, res, Dvec, yb);

  // 8) out_proj + residual: out2 = x + y @ out_proj_w^T
  gemm_bf16_wmma_kernel<<<dim3(NT / 128, DIM / 64), 256, 0, stream>>>(
      yb, D_INNER, opwb, D_INNER, NT, DIM, D_INNER, /*mode=*/3, x, out2, nullptr);

  // 9) d_out = LN(out2) + out2
  ln_add_kernel<<<NT, 256, 0, stream>>>(out2, out);
}
